// S5Dual_89653147337060
// MI455X (gfx1250) — matrix operand order repaired and checked by reading, not compiled, not run on hardware
//
#include <hip/hip_runtime.h>
#include <hip/hip_bf16.h>
#include <math.h>

// ---------------------------------------------------------------------------
// Bidirectional S5 SSM for MI455X (gfx1250, wave32, WMMA, async global->LDS).
// Pipeline:
//   prep_w1   : build f16 weight  W1[4P=1024][H=1024] = discretized B_bar (re/im, fwd/bwd)
//   prep_w2   : build f16 weight  W2[H=1024][4P=1024] = [C_re | -C_im] fwd/bwd
//   prep_lam  : Lambda_bar and Lambda_bar^512 per (dir, p)
//   f32_to_f16: x -> xh
//   gemm1     : Bu[16384][1024] f32 = xh @ W1^T          (v_wmma_f32_16x16x32_f16)
//   scan      : chunked complex scan over L (fwd ascending, bwd descending) -> xs f16
//   gemm2     : out = gelu(xs_f @ W2f^T + Df*x) + gelu(xs_b @ W2b^T + Db*x)
// GEMMs use a double-buffered GLOBAL_LOAD_ASYNC_TO_LDS_B128 pipeline when the
// toolchain exposes the builtin (ASYNCcnt-tracked), else a sync-copy fallback
// with identical buffering structure.
// ---------------------------------------------------------------------------

typedef __attribute__((ext_vector_type(16))) _Float16 v16h;
typedef __attribute__((ext_vector_type(8)))  _Float16 half8;
typedef __attribute__((ext_vector_type(4)))  _Float16 half4;
typedef __attribute__((ext_vector_type(8)))  float    v8f;

// exact parameter types of __builtin_amdgcn_global_load_async_to_lds_b128
typedef int i32x4 __attribute__((vector_size(16)));
typedef i32x4 __attribute__((address_space(1))) gl_i32x4;   // global
typedef i32x4 __attribute__((address_space(3))) ld_i32x4;   // LDS

#define NB  4
#define LSEQ 4096
#define HD  1024
#define PD  256
#define MTOT (NB * LSEQ)     // 16384

#define BM 128
#define BN 128
#define BK 32
#define BKP 40               // padded LDS row (halves) to avoid bank conflicts

#if defined(__gfx1250__) && __has_builtin(__builtin_amdgcn_global_load_async_to_lds_b128)
#define USE_ASYNC_LDS 1
#else
#define USE_ASYNC_LDS 0
#endif

// 16-byte tile chunk: global -> LDS (async on CDNA5, sync fallback otherwise)
__device__ __forceinline__ void tile_ld16(const _Float16* __restrict__ g,
                                          _Float16* l) {
#if USE_ASYNC_LDS
    __builtin_amdgcn_global_load_async_to_lds_b128(
        (gl_i32x4*)g, (ld_i32x4*)l, 0 /*imm offset*/, 0 /*cpol*/);
#else
    *(half8*)l = *(const half8*)g;
#endif
}

// wait for this wave's outstanding async tile loads, then workgroup barrier
__device__ __forceinline__ void wait_tiles() {
#if USE_ASYNC_LDS
#if __has_builtin(__builtin_amdgcn_s_wait_asynccnt)
    __builtin_amdgcn_s_wait_asynccnt(0);
#else
    asm volatile("s_wait_asynccnt 0x0" ::: "memory");
#endif
#endif
    __syncthreads();
}

// cooperative 128x32 A-tile + 128x32 B-tile load (256 threads, 4 chunks each)
__device__ __forceinline__ void load_tiles(const _Float16* __restrict__ A,
                                           const _Float16* __restrict__ W,
                                           _Float16* As, _Float16* Bs,
                                           int t, size_t aBase, size_t wBase) {
    #pragma unroll
    for (int i = 0; i < 2; ++i) {
        int chunk = t + 256 * i;
        int row = chunk >> 2;
        int kc  = chunk & 3;
        tile_ld16(&A[aBase + (size_t)row * 1024 + kc * 8], &As[row * BKP + kc * 8]);
        tile_ld16(&W[wBase + (size_t)row * 1024 + kc * 8], &Bs[row * BKP + kc * 8]);
    }
}

__device__ __forceinline__ float gelu_exact(float v) {
    return 0.5f * v * (1.0f + erff(v * 0.70710678118654752f));
}

// discretization coefficient (Lambda_bar - 1) / Lambda_safe  for channel p
__device__ __forceinline__ void disc_coef(const float* __restrict__ log_real,
                                          const float* __restrict__ imag,
                                          const float* __restrict__ log_Delta,
                                          int p, float& cr, float& ci,
                                          float& lbr, float& lbi) {
    float lr = log_real[p];
    float im = imag[p];
    float delta = expf(log_Delta[p]);
    float a = -expf(lr);                 // Re(Lambda)
    float er = expf(a * delta);
    float ang = im * delta;
    lbr = er * cosf(ang);                // Lambda_bar
    lbi = er * sinf(ang);
    float sr = a, si = im;
    if (a * a + im * im < 1e-12f) { sr = 1e-6f; si = 0.0f; }
    float den = sr * sr + si * si;
    float nr = lbr - 1.0f, ni = lbi;
    cr = (nr * sr + ni * si) / den;
    ci = (ni * sr - nr * si) / den;
}

// ---------------------------------------------------------------------------
__global__ __launch_bounds__(256) void prep_w1(
    const float* __restrict__ log_real, const float* __restrict__ imag,
    const float* __restrict__ log_Delta,
    const float* __restrict__ B_real, const float* __restrict__ B_imag,
    _Float16* __restrict__ W1, int dirOff) {
    int idx = blockIdx.x * 256 + threadIdx.x;     // P*H threads
    int p = idx >> 10;
    int h = idx & (HD - 1);
    float cr, ci, lbr, lbi;
    disc_coef(log_real, imag, log_Delta, p, cr, ci, lbr, lbi);
    float Br = B_real[p * HD + h];
    float Bi = B_imag[p * HD + h];
    W1[(size_t)(dirOff + p) * HD + h]        = (_Float16)(cr * Br - ci * Bi);
    W1[(size_t)(dirOff + PD + p) * HD + h]   = (_Float16)(cr * Bi + ci * Br);
}

__global__ __launch_bounds__(256) void prep_w2(
    const float* __restrict__ C_real, const float* __restrict__ C_imag,
    _Float16* __restrict__ W2, int dirOff) {
    int idx = blockIdx.x * 256 + threadIdx.x;     // H*P threads
    int h = idx >> 8;
    int p = idx & (PD - 1);
    float cr = C_real[h * PD + p];
    float ci = C_imag[h * PD + p];
    W2[(size_t)h * 1024 + dirOff + p]       = (_Float16)cr;
    W2[(size_t)h * 1024 + dirOff + PD + p]  = (_Float16)(-ci);
}

__global__ __launch_bounds__(256) void prep_lam(
    const float* __restrict__ log_real, const float* __restrict__ imag,
    const float* __restrict__ log_Delta,
    float2* __restrict__ lam, float2* __restrict__ lam512, int off) {
    int p = threadIdx.x;                          // 1 block, 256 threads
    float cr, ci, lbr, lbi;
    disc_coef(log_real, imag, log_Delta, p, cr, ci, lbr, lbi);
    lam[off + p] = make_float2(lbr, lbi);
    // Lambda_bar^512 = exp(512 * Lambda * Delta)
    float delta = expf(log_Delta[p]);
    float a = -expf(log_real[p]);
    float er = expf(512.0f * a * delta);
    float ang = 512.0f * imag[p] * delta;
    lam512[off + p] = make_float2(er * cosf(ang), er * sinf(ang));
}

__global__ __launch_bounds__(256) void f32_to_f16(
    const float* __restrict__ in, _Float16* __restrict__ out) {
    int i = (blockIdx.x * 256 + threadIdx.x) * 4;
    float4 v = *(const float4*)(in + i);
    half4 h;
    h.x = (_Float16)v.x; h.y = (_Float16)v.y;
    h.z = (_Float16)v.z; h.w = (_Float16)v.w;
    *(half4*)(out + i) = h;
}

// ---------------------------------------------------------------------------
// GEMM1: Bu[M][1024] f32 = xh[M][1024] @ W1[1024(n)][1024(k)]^T   (NT layout)
__global__ __launch_bounds__(256) void gemm1_kernel(
    const _Float16* __restrict__ A,   // [M][K] row-major
    const _Float16* __restrict__ W,   // [N][K] row-major
    float* __restrict__ C) {
    const int K = 1024, N = 1024;
    __shared__ __align__(16) _Float16 As[2][BM * BKP];
    __shared__ __align__(16) _Float16 Bs[2][BN * BKP];

    const int m0 = blockIdx.x * BM;
    const int n0 = blockIdx.y * BN;
    const int t = threadIdx.x;
    const int lane = t & 31, wave = t >> 5;
    const int wm = wave & 1, wn = wave >> 1;          // 2x4 wave grid
    const int mW = wm * 64, nW = wn * 32;             // wave tile 64x32
    const int r = lane & 15, kh = lane >> 4;

    v8f acc[4][2] = {};

    load_tiles(A, W, As[0], Bs[0], t, (size_t)m0 * K, (size_t)n0 * K);

    for (int kk = 0; kk < K; kk += BK) {
        const int cur = (kk >> 5) & 1;
        wait_tiles();                       // buf[cur] ready in LDS, all waves
        if (kk + BK < K)                    // overlap next tile with compute
            load_tiles(A, W, As[cur ^ 1], Bs[cur ^ 1], t,
                       (size_t)m0 * K + kk + BK, (size_t)n0 * K + kk + BK);

        union { v16h v; half8 h[2]; } af[4], bf[2];
        #pragma unroll
        for (int i = 0; i < 4; ++i) {
            int base = (mW + i * 16 + r) * BKP + 8 * kh;
            af[i].h[0] = *(const half8*)&As[cur][base];
            af[i].h[1] = *(const half8*)&As[cur][base + 16];
        }
        #pragma unroll
        for (int j = 0; j < 2; ++j) {
            int base = (nW + j * 16 + r) * BKP + 8 * kh;
            bf[j].h[0] = *(const half8*)&Bs[cur][base];
            bf[j].h[1] = *(const half8*)&Bs[cur][base + 16];
        }
        #pragma unroll
        for (int i = 0; i < 4; ++i)
            #pragma unroll
            for (int j = 0; j < 2; ++j)
                acc[i][j] = __builtin_amdgcn_wmma_f32_16x16x32_f16(
                    false, af[i].v, false, bf[j].v, (short)0, acc[i][j],
                    false, false);
    }

    // epilogue: C/D layout — VGPR v: lanes0-15 M=v, lanes16-31 M=v+8, N=lane&15
    const int rowOff = 8 * kh;
    const int colB = n0 + nW + (lane & 15);
    #pragma unroll
    for (int i = 0; i < 4; ++i)
        #pragma unroll
        for (int j = 0; j < 2; ++j)
            #pragma unroll
            for (int v = 0; v < 8; ++v) {
                int m = m0 + mW + i * 16 + v + rowOff;
                C[(size_t)m * N + colB + j * 16] = acc[i][j][v];
            }
}

// ---------------------------------------------------------------------------
// Chunked complex scan over L. Block = (b, dir, 32 channels); 8 chunks x 512.
__global__ __launch_bounds__(256) void scan_kernel(
    const float* __restrict__ Bu, const float2* __restrict__ lam,
    const float2* __restrict__ lam512, _Float16* __restrict__ xs) {
    __shared__ float2 lastS[32][8];
    __shared__ float2 initS[32][8];

    int blk = blockIdx.x;
    int b = blk >> 4;
    int rest = blk & 15;
    int dir = rest >> 3;
    int pblk = rest & 7;
    int t = threadIdx.x;
    int ch = t & 31;
    int chunk = t >> 5;
    int p = pblk * 32 + ch;

    float2 lb = lam[dir * PD + p];
    int cre = dir * 512 + p;
    int cim = cre + PD;
    long long stepRow = (dir == 0) ? 1024 : -1024;
    int l0 = (dir == 0) ? chunk * 512 : (LSEQ - 1) - chunk * 512;
    long long rowBase = ((long long)b * LSEQ + l0) * 1024;

    // Phase A: local scan, zero init; keep last state
    float sre = 0.0f, sim = 0.0f;
    long long rb = rowBase;
    for (int s = 0; s < 512; ++s) {
        float bre = Bu[rb + cre];
        float bim = Bu[rb + cim];
        float tre = lb.x * sre - lb.y * sim + bre;
        sim = lb.x * sim + lb.y * sre + bim;
        sre = tre;
        rb += stepRow;
    }
    lastS[ch][chunk] = make_float2(sre, sim);
    __syncthreads();

    // Phase B: exclusive scan across chunks (one thread per channel)
    if (chunk == 0) {
        float2 l5 = lam512[dir * PD + p];
        float car_r = 0.0f, car_i = 0.0f;
        #pragma unroll
        for (int c = 0; c < 8; ++c) {
            initS[ch][c] = make_float2(car_r, car_i);
            float2 la = lastS[ch][c];
            float nr = l5.x * car_r - l5.y * car_i + la.x;
            car_i = l5.x * car_i + l5.y * car_r + la.y;
            car_r = nr;
        }
    }
    __syncthreads();

    // Phase C: full scan with injected initial state; emit xs as f16
    float2 st = initS[ch][chunk];
    sre = st.x; sim = st.y;
    rb = rowBase;
    for (int s = 0; s < 512; ++s) {
        float bre = Bu[rb + cre];
        float bim = Bu[rb + cim];
        float tre = lb.x * sre - lb.y * sim + bre;
        sim = lb.x * sim + lb.y * sre + bim;
        sre = tre;
        xs[rb + cre] = (_Float16)sre;
        xs[rb + cim] = (_Float16)sim;
        rb += stepRow;
    }
}

// ---------------------------------------------------------------------------
// GEMM2 with fused D*x + exact GELU(fwd) + GELU(bwd) + sum epilogue.
// Xs[M][1024] (cols 0..511 fwd re/im, 512..1023 bwd), W2[H][1024] same split.
__global__ __launch_bounds__(256) void gemm2_kernel(
    const _Float16* __restrict__ Xs, const _Float16* __restrict__ W,
    const float* __restrict__ x, const float* __restrict__ Df,
    const float* __restrict__ Db, float* __restrict__ out) {
    const int K = 1024, N = 1024;
    __shared__ __align__(16) _Float16 As[2][BM * BKP];
    __shared__ __align__(16) _Float16 Bs[2][BN * BKP];

    const int m0 = blockIdx.x * BM;
    const int n0 = blockIdx.y * BN;
    const int t = threadIdx.x;
    const int lane = t & 31, wave = t >> 5;
    const int wm = wave & 1, wn = wave >> 1;
    const int mW = wm * 64, nW = wn * 32;
    const int r = lane & 15, kh = lane >> 4;

    v8f acc[2][4][2] = {};   // [dir][msub][nsub]

    load_tiles(Xs, W, As[0], Bs[0], t, (size_t)m0 * K, (size_t)n0 * K);

    for (int kk = 0; kk < K; kk += BK) {
        const int cur = (kk >> 5) & 1;
        const int d = kk >> 9;   // 0: fwd, 1: bwd
        wait_tiles();
        if (kk + BK < K)
            load_tiles(Xs, W, As[cur ^ 1], Bs[cur ^ 1], t,
                       (size_t)m0 * K + kk + BK, (size_t)n0 * K + kk + BK);

        union { v16h v; half8 h[2]; } af[4], bf[2];
        #pragma unroll
        for (int i = 0; i < 4; ++i) {
            int base = (mW + i * 16 + r) * BKP + 8 * kh;
            af[i].h[0] = *(const half8*)&As[cur][base];
            af[i].h[1] = *(const half8*)&As[cur][base + 16];
        }
        #pragma unroll
        for (int j = 0; j < 2; ++j) {
            int base = (nW + j * 16 + r) * BKP + 8 * kh;
            bf[j].h[0] = *(const half8*)&Bs[cur][base];
            bf[j].h[1] = *(const half8*)&Bs[cur][base + 16];
        }
        #pragma unroll
        for (int i = 0; i < 4; ++i)
            #pragma unroll
            for (int j = 0; j < 2; ++j)
                acc[d][i][j] = __builtin_amdgcn_wmma_f32_16x16x32_f16(
                    false, af[i].v, false, bf[j].v, (short)0, acc[d][i][j],
                    false, false);
    }

    const int rowOff = 8 * kh;
    const int colB = n0 + nW + (lane & 15);
    #pragma unroll
    for (int i = 0; i < 4; ++i)
        #pragma unroll
        for (int j = 0; j < 2; ++j) {
            int c = colB + j * 16;
            float df = Df[c], db = Db[c];
            #pragma unroll
            for (int v = 0; v < 8; ++v) {
                int m = m0 + mW + i * 16 + v + rowOff;
                float xv = x[(size_t)m * N + c];
                float yf = acc[0][i][j][v] + df * xv;
                float yb = acc[1][i][j][v] + db * xv;
                out[(size_t)m * N + c] = gelu_exact(yf) + gelu_exact(yb);
            }
        }
}

// ---------------------------------------------------------------------------
extern "C" void kernel_launch(void* const* d_in, const int* in_sizes, int n_in,
                              void* d_out, int out_size, void* d_ws, size_t ws_size,
                              hipStream_t stream) {
    const float* x = (const float*)d_in[0];
    const float* fwd_log_real  = (const float*)d_in[1];
    const float* fwd_imag      = (const float*)d_in[2];
    const float* fwd_B_real    = (const float*)d_in[3];
    const float* fwd_B_imag    = (const float*)d_in[4];
    const float* fwd_C_real    = (const float*)d_in[5];
    const float* fwd_C_imag    = (const float*)d_in[6];
    const float* fwd_D         = (const float*)d_in[7];
    const float* fwd_log_Delta = (const float*)d_in[8];
    const float* bwd_log_real  = (const float*)d_in[9];
    const float* bwd_imag      = (const float*)d_in[10];
    const float* bwd_B_real    = (const float*)d_in[11];
    const float* bwd_B_imag    = (const float*)d_in[12];
    const float* bwd_C_real    = (const float*)d_in[13];
    const float* bwd_C_imag    = (const float*)d_in[14];
    const float* bwd_D         = (const float*)d_in[15];
    const float* bwd_log_Delta = (const float*)d_in[16];
    float* out = (float*)d_out;

    // workspace layout (256B aligned slices)
    char* ws = (char*)d_ws;
    size_t off = 0;
    auto take = [&](size_t bytes) { char* p = ws + off;
                                    off = (off + bytes + 255) & ~(size_t)255;
                                    return p; };
    _Float16* xh    = (_Float16*)take((size_t)MTOT * HD * 2);        // 33.5 MB
    _Float16* W1    = (_Float16*)take((size_t)1024 * 1024 * 2);      //  2 MB
    _Float16* W2    = (_Float16*)take((size_t)1024 * 1024 * 2);      //  2 MB
    float2*   lam   = (float2*)take((size_t)512 * 8);
    float2*   lam5  = (float2*)take((size_t)512 * 8);
    float*    Bu    = (float*)take((size_t)MTOT * 1024 * 4);         // 67 MB
    _Float16* xsbuf = (_Float16*)take((size_t)MTOT * 1024 * 2);      // 33.5 MB
    (void)ws_size; (void)in_sizes; (void)n_in; (void)out_size;

    // --- parameter prep ---
    prep_w1<<<dim3(PD * HD / 256), dim3(256), 0, stream>>>(
        fwd_log_real, fwd_imag, fwd_log_Delta, fwd_B_real, fwd_B_imag, W1, 0);
    prep_w1<<<dim3(PD * HD / 256), dim3(256), 0, stream>>>(
        bwd_log_real, bwd_imag, bwd_log_Delta, bwd_B_real, bwd_B_imag, W1, 512);
    prep_w2<<<dim3(HD * PD / 256), dim3(256), 0, stream>>>(fwd_C_real, fwd_C_imag, W2, 0);
    prep_w2<<<dim3(HD * PD / 256), dim3(256), 0, stream>>>(bwd_C_real, bwd_C_imag, W2, 512);
    prep_lam<<<dim3(1), dim3(256), 0, stream>>>(
        fwd_log_real, fwd_imag, fwd_log_Delta, lam, lam5, 0);
    prep_lam<<<dim3(1), dim3(256), 0, stream>>>(
        bwd_log_real, bwd_imag, bwd_log_Delta, lam, lam5, 256);
    f32_to_f16<<<dim3((size_t)MTOT * HD / (256 * 4)), dim3(256), 0, stream>>>(x, xh);

    // --- input projection GEMM (WMMA, async double-buffered tiles) ---
    gemm1_kernel<<<dim3(MTOT / BM, 1024 / BN), dim3(256), 0, stream>>>(xh, W1, Bu);

    // --- bidirectional chunked scan ---
    scan_kernel<<<dim3(NB * 2 * 8), dim3(256), 0, stream>>>(Bu, lam, lam5, xsbuf);

    // --- output projection GEMM + fused D*x + GELU + sum (WMMA) ---
    gemm2_kernel<<<dim3(MTOT / BM, 1024 / BN), dim3(256), 0, stream>>>(
        xsbuf, W2, x, fwd_D, bwd_D, out);
}
